// VMFMixture_50672024158577
// MI455X (gfx1250) — matrix operand order repaired and checked by reading, not compiled, not run on hardware
//
#include <hip/hip_runtime.h>
#include <math.h>

#define N_PTS   262144
#define DIMS    256
#define KCLUS   256
#define BM      256
#define THREADS 512
#define NBLOCKS (N_PTS / BM)      // 1024

#define LDS_STRIDE 264            // halves per row, padded (256+8) to rotate LDS banks
#define REG0_BYTES (256 * LDS_STRIDE * 2)           // 135168 bytes per big region
#define SMEM_BYTES (2 * REG0_BYTES + 256*4 + 256*4 + 16)

typedef _Float16     v16h  __attribute__((ext_vector_type(16)));
typedef float        v8f   __attribute__((ext_vector_type(8)));
typedef unsigned int u32x4 __attribute__((ext_vector_type(4)));
typedef unsigned int u32x2 __attribute__((ext_vector_type(2)));
typedef float        f32x4 __attribute__((ext_vector_type(4)));
typedef int          i32x8 __attribute__((ext_vector_type(8)));
typedef int          i32x4 __attribute__((ext_vector_type(4)));

union V16H { v16h h; u32x4 u[2]; };
union H2U  { _Float16 f; unsigned short u; };

// ---------------------------------------------------------------------------
// Kernel: zero the output (Nk[256] | Sk[256*256] | lb[1]) every launch
// ---------------------------------------------------------------------------
__global__ void vmf_zero(float* __restrict__ p, int n) {
    int i = blockIdx.x * 256 + threadIdx.x;
    if (i < n) p[i] = 0.0f;
}

// ---------------------------------------------------------------------------
// Kernel: prep. W = f16(kappa * mu_normalized), bias = log_softmax(logpi)+logC
// ---------------------------------------------------------------------------
__global__ void vmf_prep(const float* __restrict__ mus, const float* __restrict__ kappas,
                         const float* __restrict__ logpi, unsigned short* __restrict__ Wf16,
                         float* __restrict__ bias) {
    __shared__ float red[256];
    const int t = threadIdx.x;

    // log_softmax over logpi (K=256)
    float lp = logpi[t];
    red[t] = lp; __syncthreads();
    for (int s = 128; s > 0; s >>= 1) { if (t < s) red[t] = fmaxf(red[t], red[t + s]); __syncthreads(); }
    float mx = red[0]; __syncthreads();
    red[t] = expf(lp - mx); __syncthreads();
    for (int s = 128; s > 0; s >>= 1) { if (t < s) red[t] += red[t + s]; __syncthreads(); }
    float lsep = mx + logf(red[0]);

    // normalize mu row t
    const float* mrow = mus + (size_t)t * DIMS;
    float ss = 0.0f;
    for (int d = 0; d < DIMS; ++d) { float v = mrow[d]; ss += v * v; }
    float inv = 1.0f / fmaxf(sqrtf(ss), 1e-12f);
    float kap = fmaxf(kappas[t], 1e-6f);

    // vmf logC, piecewise log I_nu (mirrors reference; double precision)
    const double nu = 0.5 * (double)DIMS - 1.0;   // 127
    double k  = (double)kap;
    double tt = fmax(k * 0.5, 1e-12);
    double base = nu * log(tt) - lgamma(nu + 1.0);
    double a1 = k * k / (4.0 * (nu + 1.0));
    double a2 = (k * k * k * k) / (32.0 * (nu + 1.0) * (nu + 2.0));
    double lsmall = base + log(fmax(1.0 + a1 + a2, 1e-12));
    double mu4  = 4.0 * nu * nu;
    double kc   = fmax(k, 1e-6), invk = 1.0 / kc;
    double c1 = -(mu4 - 1.0) * 0.125 * invk;
    double c2 = (mu4 - 1.0) * (mu4 - 9.0) * invk * invk / 128.0;
    double llarge = k - 0.5 * log(2.0 * M_PI * kc) + log(fmax(1.0 + c1 + c2, 1e-12));
    double w  = fmin(fmax((k - 2.0) / 10.0, 0.0), 1.0);
    double logIv = (k <= 2.0) ? lsmall : ((k >= 12.0) ? llarge : (1.0 - w) * lsmall + w * llarge);
    double logC  = nu * log(fmax(k, 1e-12)) - (nu + 1.0) * log(2.0 * M_PI) - logIv;

    bias[t] = (float)((double)(lp - lsep) + logC);

    // W row t in f16 (dense [K][D] row-major in workspace)
    float sc = inv * kap;
    unsigned short* wr = Wf16 + (size_t)t * DIMS;
    for (int d = 0; d < DIMS; ++d) { H2U c; c.f = (_Float16)(mrow[d] * sc); wr[d] = c.u; }
}

// ---------------------------------------------------------------------------
// Fused main kernel: G1 (logits) -> softmax -> G2 (Sk), all WMMA f16->f32.
// Block = 512 threads = 16 waves, 256 rows of X per block. W staged by TDM.
// ---------------------------------------------------------------------------
__launch_bounds__(THREADS, 1)
__global__ void vmf_main(const float* __restrict__ X, const unsigned short* __restrict__ Wf16,
                         const float* __restrict__ bias, float* __restrict__ out) {
    extern __shared__ char smem[];
    unsigned short* sXn  = (unsigned short*)smem;                  // [256 rows][264] f16 (Xn)
    unsigned short* sW   = (unsigned short*)(smem + REG0_BYTES);   // [256 k   ][264] f16 (W)
    unsigned short* gT   = sXn;                                    // reused: gamma^T [k][i]
    unsigned short* sXnT = sW;                                     // reused: Xn^T   [d][i]
    float* sBias = (float*)(smem + 2 * REG0_BYTES);                // [256]
    float* sNk   = sBias + 256;                                    // [256]
    float* sLb   = sNk + 256;                                      // [1]

    const int  t    = threadIdx.x;
    const int  wave = t >> 5;
    const int  lane = t & 31;
    const int  l16  = lane & 15;
    const bool hi   = lane >= 16;
    const int  i0   = wave * 16;                       // block-local row base for this wave
    const size_t gr0 = (size_t)blockIdx.x * BM;

    // -------- phase 1a: kick off TDM load of W (global, dense) -> LDS (padded) --------
    // 2D tile: 256 x 256 f16, tensor stride 256; LDS padding: after every
    // 128 DWORDs (512B row) insert 4 DWORDs (16B) -> 264-half LDS stride.
    if (wave == 0) {
        unsigned long long ga = (unsigned long long)(uintptr_t)Wf16;
        unsigned lds_off = (unsigned)(uintptr_t)(smem + REG0_BYTES);   // flat addr low32 = LDS offset

        u32x4 g0;
        g0[0] = 1u;                                    // count=1, user descriptor
        g0[1] = lds_off;                               // lds_addr (bytes)
        g0[2] = (unsigned)(ga & 0xffffffffu);          // global_addr[31:0]
        g0[3] = (unsigned)((ga >> 32) & 0x1ffffffu)    // global_addr[56:32]
              | (2u << 30);                            // type = 2 ("image")

        i32x8 g1;
        g1[0] = (int)((1u << 16)                       // data_size = 1 (2 bytes)
              |       (1u << 20)                       // pad_enable
              |       (6u << 22)                       // pad_interval: 2^(6+1)=128 DWORDs
              |       (3u << 25));                     // pad_amount: 3+1 = 4 DWORDs (16B)
        g1[1] = (int)(256u << 16);                     // tensor_dim0 = 256 (bits 79:48)
        g1[2] = (int)(256u << 16);                     // dim0 hi=0 | tensor_dim1 = 256 (bits 111:80)
        g1[3] = (int)(256u << 16);                     // dim1 hi=0 | tile_dim0 = 256 (bits 127:112)
        g1[4] = (int)256u;                             // tile_dim1 = 256 | tile_dim2 = 0
        g1[5] = (int)256u;                             // tensor_dim0_stride = 256 (bits 207:160)
        g1[6] = 0;                                     // stride0 hi | tensor_dim1_stride lo
        g1[7] = 0;

        i32x4 gz4; gz4[0] = 0; gz4[1] = 0; gz4[2] = 0; gz4[3] = 0;   // groups 2/3: 2D tensor
        i32x8 gz8;
        #pragma unroll
        for (int e = 0; e < 8; ++e) gz8[e] = 0;
        // clang-23 / therock 6-arg form: (v4u, v8i, v4i, v4i, v8i, i32 cpol)
        __builtin_amdgcn_tensor_load_to_lds(g0, g1, gz4, gz4, gz8, 0);
    }

    // -------- phase 1b: bias/Nk init + X load/normalize -> f16 LDS --------
    if (t < 256) { sBias[t] = bias[t]; sNk[t] = 0.0f; }
    if (t == 0)  { sLb[0] = 0.0f; }

    {   // X: two threads per row; row-norm via partner shuffle; f16 into LDS
        const int rowl = t >> 1;
        const int c0   = (t & 1) * 128;
        const f32x4* px = (const f32x4*)(X + (gr0 + rowl) * DIMS + c0);  // 32 x f32x4
        float ss = 0.0f;
        #pragma unroll
        for (int j = 0; j < 32; ++j) {
            f32x4 v = px[j];
            ss += v.x * v.x + v.y * v.y + v.z * v.z + v.w * v.w;
        }
        ss += __shfl_xor(ss, 1, 32);
        const float sc = 1.0f / fmaxf(sqrtf(ss), 1e-12f);
        unsigned short* dst = sXn + rowl * LDS_STRIDE + c0;
        #pragma unroll
        for (int j = 0; j < 32; ++j) {
            f32x4 v = px[j];
            H2U h0, h1, h2, h3;
            h0.f = (_Float16)(v.x * sc); h1.f = (_Float16)(v.y * sc);
            h2.f = (_Float16)(v.z * sc); h3.f = (_Float16)(v.w * sc);
            u32x2 pk;
            pk.x = (unsigned)h0.u | ((unsigned)h1.u << 16);
            pk.y = (unsigned)h2.u | ((unsigned)h3.u << 16);
            *(u32x2*)(dst + j * 4) = pk;
        }
    }
    if (wave == 0) __builtin_amdgcn_s_wait_tensorcnt(0);   // W tile landed in LDS
    __syncthreads();

    // -------- phase 2: G1  logits[16 x 256] per wave, K-dim = D in chunks of 32 --------
    v8f vz;
    #pragma unroll
    for (int e = 0; e < 8; ++e) vz[e] = 0.0f;
    v8f acc[16];
    #pragma unroll
    for (int kt = 0; kt < 16; ++kt) acc[kt] = vz;

    {
        const int arow = i0 + l16;
        #pragma unroll
        for (int dc = 0; dc < 8; ++dc) {
            V16H a;   // A 16x32 f16: lane<16 holds K 0..7,16..23 ; lane>=16 holds 8..15,24..31
            const unsigned short* pa = sXn + arow * LDS_STRIDE + dc * 32 + (hi ? 8 : 0);
            a.u[0] = *(const u32x4*)(pa);
            a.u[1] = *(const u32x4*)(pa + 16);
            #pragma unroll
            for (int kt = 0; kt < 16; ++kt) {
                V16H b;  // B 32x16 f16: lane = column k; lane<16 holds d 0..7,16..23 ; lane>=16 holds 8..15,24..31
                const unsigned short* pb = sW + (kt * 16 + l16) * LDS_STRIDE + dc * 32 + (hi ? 8 : 0);
                b.u[0] = *(const u32x4*)(pb);
                b.u[1] = *(const u32x4*)(pb + 16);
                acc[kt] = __builtin_amdgcn_wmma_f32_16x16x32_f16(
                    false, a.h, false, b.h, (short)0, acc[kt], false, false);
            }
        }
    }
    __syncthreads();

    // -------- phase 3a: transpose Xn -> XnT (old W region now dead) --------
    for (int it = 0; it < 128; ++it) {
        int idx = t + it * THREADS;                    // 0..65535
        int d = idx & 255, i = idx >> 8;
        sXnT[d * LDS_STRIDE + i] = sXn[i * LDS_STRIDE + d];
    }
    __syncthreads();

    // -------- phase 3b: softmax over K, gamma(f16) -> gT (old Xn region), Nk, lb --------
    {
        float bk[16];
        #pragma unroll
        for (int kt = 0; kt < 16; ++kt) bk[kt] = sBias[kt * 16 + l16];

        float lse[8];
        float lbsum = 0.0f;
        #pragma unroll
        for (int r = 0; r < 8; ++r) {
            float m = -3.4e38f;
            #pragma unroll
            for (int kt = 0; kt < 16; ++kt) m = fmaxf(m, acc[kt][r] + bk[kt]);
            #pragma unroll
            for (int j = 1; j < 16; j <<= 1) m = fmaxf(m, __shfl_xor(m, j, 16));
            float s = 0.0f;
            #pragma unroll
            for (int kt = 0; kt < 16; ++kt) s += __expf(acc[kt][r] + bk[kt] - m);
            #pragma unroll
            for (int j = 1; j < 16; j <<= 1) s += __shfl_xor(s, j, 16);
            lse[r] = m + __logf(s);
            if (l16 == 0) lbsum += lse[r];
        }
        #pragma unroll
        for (int kt = 0; kt < 16; ++kt) {
            float nk = 0.0f;
            union { u32x4 u; unsigned short us[8]; } pk;
            #pragma unroll
            for (int r = 0; r < 8; ++r) {
                float g = __expf(acc[kt][r] + bk[kt] - lse[r]);
                nk += g;
                H2U c; c.f = (_Float16)g; pk.us[r] = c.u;
            }
            int k = kt * 16 + l16;
            *(u32x4*)(gT + k * LDS_STRIDE + i0 + (hi ? 8 : 0)) = pk.u;
            atomicAdd(&sNk[k], nk);
        }
        if (l16 == 0) atomicAdd(&sLb[0], lbsum);
    }
    __syncthreads();

    // -------- phase 4: G2  Sk[16k x 256d] per wave = gamma^T x Xn --------
    // ic outer so each gamma^T A-tile is loaded once and reused across all 16 d-tiles.
    {
        v8f acc2[4][4];                                // [dp][dt] = 128 VGPRs
        #pragma unroll
        for (int dp = 0; dp < 4; ++dp)
            #pragma unroll
            for (int dt = 0; dt < 4; ++dt) acc2[dp][dt] = vz;

        const int kbase = wave * 16;
        #pragma unroll
        for (int ic = 0; ic < 8; ++ic) {               // i chunks of 32
            V16H a;                                    // A = gamma^T tile 16k x 32i
            const unsigned short* pa = gT + (kbase + l16) * LDS_STRIDE + ic * 32 + (hi ? 8 : 0);
            a.u[0] = *(const u32x4*)(pa);
            a.u[1] = *(const u32x4*)(pa + 16);
            #pragma unroll
            for (int dp = 0; dp < 4; ++dp) {
                #pragma unroll
                for (int dt = 0; dt < 4; ++dt) {       // B = Xn tile 32i x 16d (from XnT)
                    V16H b;
                    const unsigned short* pb =
                        sXnT + (dp * 64 + dt * 16 + l16) * LDS_STRIDE + ic * 32 + (hi ? 8 : 0);
                    b.u[0] = *(const u32x4*)(pb);
                    b.u[1] = *(const u32x4*)(pb + 16);
                    acc2[dp][dt] = __builtin_amdgcn_wmma_f32_16x16x32_f16(
                        false, a.h, false, b.h, (short)0, acc2[dp][dt], false, false);
                }
            }
        }
        // atomic flush: each block contributes each Sk element exactly once
        float* outSk = out + 256;
        #pragma unroll
        for (int dp = 0; dp < 4; ++dp) {
            #pragma unroll
            for (int dt = 0; dt < 4; ++dt) {
                int d = dp * 64 + dt * 16 + l16;
                #pragma unroll
                for (int r = 0; r < 8; ++r) {
                    int k = kbase + (hi ? 8 + r : r);
                    atomicAdd(&outSk[k * 256 + d], acc2[dp][dt][r]);
                }
            }
        }
    }
    __syncthreads();

    // -------- phase 5: block-level Nk / lb flush --------
    if (t < 256) atomicAdd(&out[t], sNk[t]);
    if (t == 0)  atomicAdd(&out[256 + KCLUS * DIMS], sLb[0]);
}

// ---------------------------------------------------------------------------
extern "C" void kernel_launch(void* const* d_in, const int* in_sizes, int n_in,
                              void* d_out, int out_size, void* d_ws, size_t ws_size,
                              hipStream_t stream) {
    const float* X      = (const float*)d_in[0];
    const float* mus    = (const float*)d_in[1];
    const float* kappas = (const float*)d_in[2];
    const float* logpi  = (const float*)d_in[3];
    float* out = (float*)d_out;

    unsigned short* Wf16 = (unsigned short*)d_ws;                      // 128 KB
    float* bias = (float*)((char*)d_ws + (size_t)KCLUS * DIMS * 2);    // +1 KB

    (void)in_sizes; (void)n_in; (void)ws_size;

    (void)hipFuncSetAttribute((const void*)vmf_main,
                              hipFuncAttributeMaxDynamicSharedMemorySize, SMEM_BYTES);

    vmf_zero<<<(out_size + 255) / 256, 256, 0, stream>>>(out, out_size);
    vmf_prep<<<1, 256, 0, stream>>>(mus, kappas, logpi, Wf16, bias);
    vmf_main<<<NBLOCKS, THREADS, SMEM_BYTES, stream>>>(X, Wf16, bias, out);
}
